// MultiHeadAttention_2619930051748
// MI455X (gfx1250) — matrix operand order repaired and checked by reading, not compiled, not run on hardware
//
#include <hip/hip_runtime.h>

#define DEVI __device__ __forceinline__

typedef __attribute__((ext_vector_type(16))) __bf16 v16bf;
typedef __attribute__((ext_vector_type(8)))  float  v8f;
typedef __attribute__((ext_vector_type(4)))  int    v4i;

static constexpr int TT = 2048;   // seq len
static constexpr int CC = 1024;   // d_model
static constexpr int HH = 16;     // heads
static constexpr int HD = 64;     // head dim
static constexpr int BB = 4;      // batch
static constexpr int MR = BB * TT;                        // 8192 rows
static constexpr size_t QSZ = (size_t)BB * HH * TT * HD;  // elems per q/k/vT buffer

// ---------- bf16 helpers ----------
DEVI unsigned short f2bf(float f) {
    unsigned int x = __builtin_bit_cast(unsigned int, f);
    unsigned int r = x + 0x7FFFu + ((x >> 16) & 1u);
    return (unsigned short)(r >> 16);
}
DEVI unsigned int pack2bf(float lo, float hi) {
    return (unsigned int)f2bf(lo) | ((unsigned int)f2bf(hi) << 16);
}

// ---------- async global -> LDS (gfx1250, ASYNCcnt-tracked DMA) ----------
DEVI void async_b128(const void* g, void* l) {
    __builtin_amdgcn_global_load_async_to_lds_b128(
        (__attribute__((address_space(1))) v4i*)g,
        (__attribute__((address_space(3))) v4i*)l, 0, 0);
}
#define WAIT_ASYNC(n) asm volatile("s_wait_asynccnt " #n ::: "memory")

// ---------- WMMA fragment loaders (dword-granular, K-major source) ----------
// A operand (16x32 bf16): lane m = lane&15, half hh = lane>>4.
// VGPR v holds kpair (v&3) + (v>=4 ? 8:0) + hh*4   (kpair = 2 consecutive K)
DEVI v16bf ldFragA(const unsigned int* base, int strideDw, int row, int hh, int kpofs) {
    union { unsigned int u[8]; v16bf v; } f;
#pragma unroll
    for (int i = 0; i < 8; ++i) {
        int kp = (i & 3) + ((i >> 2) << 3) + hh * 4 + kpofs;
        f.u[i] = base[row * strideDw + kp];
    }
    return f.v;
}
// B operand (32x16 bf16): lane n = lane&15. VGPR v holds kpair (v&3) + (v>=4 ? 8:0) + hh*4.
DEVI v16bf ldFragB(const unsigned int* base, int strideDw, int row, int hh, int kpofs) {
    union { unsigned int u[8]; v16bf v; } f;
#pragma unroll
    for (int i = 0; i < 8; ++i)
        f.u[i] = base[row * strideDw + (i & 3) + ((i >> 2) << 3) + hh * 4 + kpofs];
    return f.v;
}

DEVI v8f wmma_bf16(v16bf a, v16bf b, v8f c) {
    return __builtin_amdgcn_wmma_f32_16x16x32_bf16(false, a, false, b, (short)0, c,
                                                   false, false);
}
DEVI v8f zero8() { v8f z = {0.f, 0.f, 0.f, 0.f, 0.f, 0.f, 0.f, 0.f}; return z; }

// ============================================================================
// GEMM: C[M,N] = A[M,K] @ W[N,K]^T + bias[N]
//   EPI==0: write fp32 C               (output projection)
//   EPI==1: scatter bf16 q/k/vT        (QKV projection)
//   AF32==1: A is fp32 (VGPR convert path), else A is bf16 (async DMA path)
// Block tile 128x64, 8 waves of 32x32, K-step 32.
// ============================================================================
template <int EPI, int AF32>
__global__ __launch_bounds__(256) void gemm_bf16_wmma(
    const void* __restrict__ Aptr, const float* __restrict__ W,
    const float* __restrict__ bias, float* __restrict__ Cout,
    unsigned short* __restrict__ qo, unsigned short* __restrict__ ko,
    unsigned short* __restrict__ vto, int M, int N, int K) {
    __shared__ __align__(16) unsigned int As[128 * 16];  // [row][kpair]
    __shared__ __align__(16) unsigned int Bs[64 * 16];

    const int tid = threadIdx.x;
    const int lane = tid & 31;
    const int wv = tid >> 5;
    const int wm = (wv >> 1) * 32;
    const int wn = (wv & 1) * 32;
    const int mm = lane & 15;
    const int hh = lane >> 4;

    const int n0 = blockIdx.x * 64;
    const int m0 = blockIdx.y * 128;

    v8f acc[2][2];
#pragma unroll
    for (int i = 0; i < 2; ++i)
#pragma unroll
        for (int j = 0; j < 2; ++j) acc[i][j] = zero8();

    for (int k0 = 0; k0 < K; k0 += 32) {
        __syncthreads();
        if constexpr (AF32) {
            const float* A = (const float*)Aptr;
#pragma unroll
            for (int p = 0; p < 4; ++p) {
                int r = (tid >> 3) + p * 32;
                int c = (tid & 7) * 4;
                float4 f = *(const float4*)(A + (size_t)(m0 + r) * K + k0 + c);
                As[r * 16 + (c >> 1)] = pack2bf(f.x, f.y);
                As[r * 16 + (c >> 1) + 1] = pack2bf(f.z, f.w);
            }
        } else {
            // bf16 A: straight copy -> async DMA to LDS (no VGPR round trip)
            const unsigned short* A = (const unsigned short*)Aptr;
            int r = tid >> 1;
            int sg = (tid & 1) * 2;
            async_b128(A + (size_t)(m0 + r) * K + k0 + sg * 8, &As[r * 16 + sg * 4]);
            async_b128(A + (size_t)(m0 + r) * K + k0 + (sg + 1) * 8,
                       &As[r * 16 + (sg + 1) * 4]);
        }
#pragma unroll
        for (int p = 0; p < 2; ++p) {
            int r = (tid >> 3) + p * 32;
            int c = (tid & 7) * 4;
            float4 f = *(const float4*)(W + (size_t)(n0 + r) * K + k0 + c);
            Bs[r * 16 + (c >> 1)] = pack2bf(f.x, f.y);
            Bs[r * 16 + (c >> 1) + 1] = pack2bf(f.z, f.w);
        }
        if constexpr (!AF32) WAIT_ASYNC(0x0);
        __syncthreads();

        v16bf af[2], bf[2];
#pragma unroll
        for (int i = 0; i < 2; ++i) af[i] = ldFragA(As, 16, wm + i * 16 + mm, hh, 0);
#pragma unroll
        for (int j = 0; j < 2; ++j) bf[j] = ldFragB(Bs, 16, wn + j * 16 + mm, hh, 0);
#pragma unroll
        for (int i = 0; i < 2; ++i)
#pragma unroll
            for (int j = 0; j < 2; ++j) acc[i][j] = wmma_bf16(af[i], bf[j], acc[i][j]);
    }

    // Epilogue. D layout: row = v + hh*8, col = lane&15.
#pragma unroll
    for (int i = 0; i < 2; ++i) {
#pragma unroll
        for (int j = 0; j < 2; ++j) {
#pragma unroll
            for (int v = 0; v < 8; ++v) {
                int gr = m0 + wm + i * 16 + v + hh * 8;
                int gc = n0 + wn + j * 16 + mm;
                float val = acc[i][j][v] + bias[gc];
                if constexpr (EPI == 0) {
                    Cout[(size_t)gr * N + gc] = val;
                } else {
                    int sec = gc >> 10;  // 0:q 1:k 2:v
                    int cc = gc & 1023;
                    int h = cc >> 6;
                    int d = cc & 63;
                    int b = gr >> 11;  // / TT
                    int t = gr & 2047;
                    unsigned short bv = f2bf(val);
                    size_t bh = (size_t)b * HH + h;
                    if (sec == 0)
                        qo[(bh * TT + t) * HD + d] = bv;
                    else if (sec == 1)
                        ko[(bh * TT + t) * HD + d] = bv;
                    else
                        vto[(bh * HD + d) * TT + t] = bv;  // V transposed
                }
            }
        }
    }
}

// ============================================================================
// Flash attention: one block = (b, h, 128 queries). 8 waves x 16 query rows.
// Double-buffered async-DMA K/V^T staging; online softmax in registers.
// ============================================================================
__global__ __launch_bounds__(256) void attn_flash(
    const unsigned short* __restrict__ q, const unsigned short* __restrict__ kx,
    const unsigned short* __restrict__ vt, const int* __restrict__ pad,
    unsigned short* __restrict__ aout) {
    __shared__ __align__(16) unsigned int Kt[2][64 * 32];    // [buf][key][d-pair]
    __shared__ __align__(16) unsigned int Vt[2][64 * 32];    // [buf][d][key-pair]
    __shared__ __align__(16) unsigned short Pb[8][16 * 64];  // per-wave P tile

    const int tid = threadIdx.x;
    const int lane = tid & 31;
    const int w = tid >> 5;
    const int mm = lane & 15;
    const int hh = lane >> 4;

    const int b = blockIdx.z;
    const int h = blockIdx.y;
    const int q0 = blockIdx.x * 128;
    const size_t bh = (size_t)b * HH + h;

    const unsigned short* kbase = kx + bh * TT * HD;
    const unsigned short* vbase = vt + bh * HD * TT;
    const int* pmb = pad + b * TT;
    const float scale = 0.125f;  // 1/sqrt(64)

    // per-thread staging addresses: 4 async B128s per thread per chunk
    const int sr = tid >> 2;       // 0..63 : K row (key) / V row (d)
    const int sg = (tid & 3) * 2;  // uint4 index within row

    // prologue: kick off chunk 0 into buffer 0
    {
        const unsigned short* ks = kbase + (size_t)sr * HD;
        const unsigned short* vs = vbase + (size_t)sr * TT;
        async_b128(ks + sg * 8, &Kt[0][sr * 32 + sg * 4]);
        async_b128(ks + (sg + 1) * 8, &Kt[0][sr * 32 + (sg + 1) * 4]);
        async_b128(vs + sg * 8, &Vt[0][sr * 32 + sg * 4]);
        async_b128(vs + (sg + 1) * 8, &Vt[0][sr * 32 + (sg + 1) * 4]);
    }

    // Q fragments for this wave's 16 rows (held for the whole kernel)
    const unsigned int* qrow = (const unsigned int*)(q + (bh * TT + q0 + w * 16) * HD);
    v16bf qa0 = ldFragA(qrow, 32, mm, hh, 0);   // d 0..31
    v16bf qa1 = ldFragA(qrow, 32, mm, hh, 16);  // d 32..63

    v8f accO[4];
#pragma unroll
    for (int j = 0; j < 4; ++j) accO[j] = zero8();
    float mrow[8], lrow[8];
#pragma unroll
    for (int v = 0; v < 8; ++v) { mrow[v] = -1e30f; lrow[v] = 0.f; }

    constexpr int NCH = TT / 64;  // 32 chunks
    for (int it = 0; it < NCH; ++it) {
        const int cur = it & 1;
        const int kc = it * 64;
        __syncthreads();  // all waves done reading buffer cur^1 (chunk it-1)
        if (it + 1 < NCH) {
            // prefetch chunk it+1 into the other buffer (async DMA)
            const unsigned short* ks = kbase + (size_t)(kc + 64 + sr) * HD;
            const unsigned short* vs = vbase + (size_t)sr * TT + kc + 64;
            unsigned int* kd = &Kt[cur ^ 1][sr * 32];
            unsigned int* vd = &Vt[cur ^ 1][sr * 32];
            async_b128(ks + sg * 8, kd + sg * 4);
            async_b128(ks + (sg + 1) * 8, kd + (sg + 1) * 4);
            async_b128(vs + sg * 8, vd + sg * 4);
            async_b128(vs + (sg + 1) * 8, vd + (sg + 1) * 4);
            WAIT_ASYNC(0x4);  // async loads retire in order: chunk `it` landed
        } else {
            WAIT_ASYNC(0x0);
        }
        __syncthreads();  // every wave has waited -> whole chunk visible

        // S = Q K^T  (4 key sub-tiles of 16)
        v8f s[4];
#pragma unroll
        for (int j = 0; j < 4; ++j) {
            v16bf kb0 = ldFragB(Kt[cur], 32, j * 16 + mm, hh, 0);
            v16bf kb1 = ldFragB(Kt[cur], 32, j * 16 + mm, hh, 16);
            v8f z = zero8();
            z = wmma_bf16(qa0, kb0, z);
            z = wmma_bf16(qa1, kb1, z);
            s[j] = z;
        }
        // scale + pad mask (additive -1e30 on masked keys)
        float madd[4];
#pragma unroll
        for (int j = 0; j < 4; ++j)
            madd[j] = pmb[kc + j * 16 + mm] ? -1e30f : 0.0f;
#pragma unroll
        for (int j = 0; j < 4; ++j)
#pragma unroll
            for (int v = 0; v < 8; ++v) s[j][v] = s[j][v] * scale + madd[j];

        // online softmax: row (v + hh*8) lives across 16 lanes of this half
#pragma unroll
        for (int v = 0; v < 8; ++v) {
            float cm = fmaxf(fmaxf(s[0][v], s[1][v]), fmaxf(s[2][v], s[3][v]));
#pragma unroll
            for (int off = 1; off < 16; off <<= 1)
                cm = fmaxf(cm, __shfl_xor(cm, off, 32));
            float mnew = fmaxf(mrow[v], cm);
            float sf = __expf(mrow[v] - mnew);
            mrow[v] = mnew;
            float rs = 0.f;
#pragma unroll
            for (int j = 0; j < 4; ++j) {
                float p = __expf(s[j][v] - mnew);
                s[j][v] = p;
                rs += p;
            }
#pragma unroll
            for (int off = 1; off < 16; off <<= 1) rs += __shfl_xor(rs, off, 32);
            lrow[v] = lrow[v] * sf + rs;
#pragma unroll
            for (int j = 0; j < 4; ++j) accO[j][v] *= sf;
        }
        // spill P (D layout) -> LDS row-major so it can re-enter as A fragments
#pragma unroll
        for (int j = 0; j < 4; ++j)
#pragma unroll
            for (int v = 0; v < 8; ++v)
                Pb[w][(v + hh * 8) * 64 + j * 16 + mm] = f2bf(s[j][v]);
        __syncthreads();

        // O += P V   (4 d sub-tiles, contraction over 64 keys)
        const unsigned int* Pw = (const unsigned int*)Pb[w];
        v16bf pa0 = ldFragA(Pw, 32, mm, hh, 0);   // keys 0..31
        v16bf pa1 = ldFragA(Pw, 32, mm, hh, 16);  // keys 32..63
#pragma unroll
        for (int j = 0; j < 4; ++j) {
            v16bf vb0 = ldFragB(Vt[cur], 32, j * 16 + mm, hh, 0);
            v16bf vb1 = ldFragB(Vt[cur], 32, j * 16 + mm, hh, 16);
            accO[j] = wmma_bf16(pa0, vb0, accO[j]);
            accO[j] = wmma_bf16(pa1, vb1, accO[j]);
        }
    }

    // normalize and store bf16 attn output [B*T, C]
#pragma unroll
    for (int v = 0; v < 8; ++v) {
        float inv = 1.0f / lrow[v];
        size_t gr = (size_t)b * TT + q0 + w * 16 + v + hh * 8;
#pragma unroll
        for (int j = 0; j < 4; ++j) {
            int gc = h * HD + j * 16 + mm;
            aout[gr * CC + gc] = f2bf(accO[j][v] * inv);
        }
    }
}

// ============================================================================
extern "C" void kernel_launch(void* const* d_in, const int* in_sizes, int n_in,
                              void* d_out, int out_size, void* d_ws, size_t ws_size,
                              hipStream_t stream) {
    const float* x = (const float*)d_in[0];
    const int* pad_mask = (const int*)d_in[1];
    const float* w_qkv = (const float*)d_in[2];
    const float* b_qkv = (const float*)d_in[3];
    const float* w_proj = (const float*)d_in[4];
    const float* b_proj = (const float*)d_in[5];
    float* out = (float*)d_out;

    unsigned short* qws = (unsigned short*)d_ws;
    unsigned short* kws = qws + QSZ;
    unsigned short* vtws = kws + QSZ;
    unsigned short* aout = vtws + QSZ;  // [8192,1024] bf16

    // QKV projection: [8192,3072] = x[8192,1024] @ w_qkv^T, scatter q/k/vT bf16
    gemm_bf16_wmma<1, 1><<<dim3(3 * CC / 64, MR / 128), 256, 0, stream>>>(
        (const void*)x, w_qkv, b_qkv, nullptr, qws, kws, vtws, MR, 3 * CC, CC);

    // flash attention per (b, h, 128-query tile)
    attn_flash<<<dim3(TT / 128, HH, BB), 256, 0, stream>>>(qws, kws, vtws, pad_mask,
                                                           aout);

    // output projection: out[8192,1024] = attn @ w_proj^T + b_proj (fp32 out)
    gemm_bf16_wmma<0, 0><<<dim3(CC / 64, MR / 128), 256, 0, stream>>>(
        (const void*)aout, w_proj, b_proj, out, nullptr, nullptr, nullptr, MR, CC, CC);
}
